// GraphAttentionLayer_61314953117892
// MI455X (gfx1250) — matrix operand order repaired and checked by reading, not compiled, not run on hardware
//
#include <hip/hip_runtime.h>
#include <stdint.h>

#define B_    8
#define N_    2048
#define DIN_  256
#define DOUT_ 256
#define H_    8
#define HD_   32
#define SCALE_ 0.17677669529663687f   // 32^-0.5

typedef _Float16 v16h __attribute__((ext_vector_type(16)));
typedef _Float16 v8h  __attribute__((ext_vector_type(8)));
typedef float    v8f  __attribute__((ext_vector_type(8)));

// ---------------------------------------------------------------------------
// WMMA fragment loaders per CDNA5 ISA layouts (CDNA5 ISA §7.12.2), wave32.
// A (16x32 f16, row-major (m,k) source): lane L: row m=L%16, hi=L/16,
//   v0..3 hold K = hi*8 + 0..7, v4..7 hold K = 16 + hi*8 + 0..7
//   -> two contiguous 16B segments.
// B (32x16 f16): element B[k][n]; source is B^T row-major (n,k): lane L:
//   col n=L%16, hi=L/16, v0..3 hold K = hi*8 + 0..7, v4..7 hold K = 16 + hi*8 + 0..7.
// ---------------------------------------------------------------------------
__device__ __forceinline__ v16h load_frag_A(const _Float16* base, int ld, int lane) {
  int m = lane & 15, hi = lane >> 4;
  v8h a = *(const v8h*)(base + m * ld + hi * 8);
  v8h b = *(const v8h*)(base + m * ld + 16 + hi * 8);
  v16h f;
#pragma unroll
  for (int i = 0; i < 8; ++i) { f[i] = a[i]; f[i + 8] = b[i]; }
  return f;
}

__device__ __forceinline__ v16h load_frag_B(const _Float16* base, int ld, int lane) {
  int n = lane & 15, hi = lane >> 4;
  v8h a = *(const v8h*)(base + n * ld + hi * 8);
  v8h b = *(const v8h*)(base + n * ld + 16 + hi * 8);
  v16h f;
#pragma unroll
  for (int i = 0; i < 8; ++i) { f[i] = a[i]; f[i + 8] = b[i]; }
  return f;
}

__device__ __forceinline__ v8f wmma16(v16h a, v16h b, v8f c) {
  return __builtin_amdgcn_wmma_f32_16x16x32_f16(false, a, false, b, (short)0, c,
                                                false, false);
}

// ---------------------------------------------------------------------------
// Phase 0a: f32 -> f16 conversion
// ---------------------------------------------------------------------------
__global__ __launch_bounds__(256) void cvt_f16_kernel(const float* __restrict__ s,
                                                      _Float16* __restrict__ d, int n) {
  int i = blockIdx.x * blockDim.x + threadIdx.x;
  if (i < n) d[i] = (_Float16)s[i];
}

// ---------------------------------------------------------------------------
// Phase 0b: pack adjacency int32 -> bitmask (1 bit per edge).
// Each wave produces 32 consecutive words via __ballot.
// ---------------------------------------------------------------------------
__global__ __launch_bounds__(256) void pack_adj_kernel(const int* __restrict__ adj,
                                                       uint32_t* __restrict__ bits) {
  int wave = (blockIdx.x * blockDim.x + threadIdx.x) >> 5;
  int lane = threadIdx.x & 31;
  int base = wave * 1024;  // 32 words * 32 elements
#pragma unroll 4
  for (int w = 0; w < 32; ++w) {
    int v = adj[base + w * 32 + lane];
    unsigned long long m = __ballot(v != 0);
    if (lane == 0) bits[wave * 32 + w] = (uint32_t)m;
  }
}

// ---------------------------------------------------------------------------
// Phase 1: Q/K/V projections. One wave computes a 16x16 tile of Y = x W^T + b.
// z=0 -> Qh [bh][n][32], z=1 -> Kh [bh][n][32], z=2 -> Vt [bh][32][n] (transposed)
// ---------------------------------------------------------------------------
__global__ __launch_bounds__(32) void proj_kernel(
    const _Float16* __restrict__ xh, const _Float16* __restrict__ Wqh,
    const _Float16* __restrict__ Wkh, const _Float16* __restrict__ Wvh,
    const float* __restrict__ bq, const float* __restrict__ bk,
    const float* __restrict__ bv, _Float16* __restrict__ Qh,
    _Float16* __restrict__ Kh, _Float16* __restrict__ Vt) {
  int lane = threadIdx.x & 31;
  int rowblk = blockIdx.x;  // 0..1023 (16384 rows / 16)
  int ot = blockIdx.y;      // 0..15  (256 cols / 16)
  int z = blockIdx.z;       // 0..2
  const _Float16* W = (z == 0) ? Wqh : (z == 1) ? Wkh : Wvh;
  const float* bias = (z == 0) ? bq : (z == 1) ? bk : bv;

  const _Float16* Abase = xh + (size_t)rowblk * 16 * DIN_;
  const _Float16* Bbase = W + (size_t)ot * 16 * DIN_;

  v8f c = {0.f, 0.f, 0.f, 0.f, 0.f, 0.f, 0.f, 0.f};
#pragma unroll
  for (int kc = 0; kc < DIN_ / 32; ++kc) {
    v16h a = load_frag_A(Abase + kc * 32, DIN_, lane);
    v16h b = load_frag_B(Bbase + kc * 32, DIN_, lane);
    c = wmma16(a, b, c);
  }

  int hi = lane >> 4;
  int o = ot * 16 + (lane & 15);
  float bs = bias[o];
  int bn0 = rowblk * 16;
  int bidx = bn0 >> 11;  // batch (16-row blocks never cross batch)
  int h = o >> 5, d = o & 31;
#pragma unroll
  for (int r = 0; r < 8; ++r) {
    int n = (bn0 + r + 8 * hi) & (N_ - 1);
    float y = c[r] + bs;
    if (z == 2) {
      Vt[(((size_t)(bidx * H_ + h)) * HD_ + d) * N_ + n] = (_Float16)y;
    } else {
      _Float16* dst = (z == 0) ? Qh : Kh;
      dst[((size_t)(bidx * H_ + h) * N_ + n) * HD_ + d] = (_Float16)y;
    }
  }
}

// ---------------------------------------------------------------------------
// Phase 2: masked flash attention. Block = 4 waves = 64 Q rows per (b,h).
// K/V tiles (32 keys) are staged into LDS double buffers and shared by all
// 4 waves (4x cut in K/V read traffic vs one-wave blocks). Software-pipelined:
// next chunk's global loads are issued before the current chunk's compute.
// ---------------------------------------------------------------------------
__global__ __launch_bounds__(128) void attn_kernel(
    const _Float16* __restrict__ Qh, const _Float16* __restrict__ Kh,
    const _Float16* __restrict__ Vt, const uint32_t* __restrict__ maskbits,
    _Float16* __restrict__ Oh) {
  __shared__ uint32_t mlds[64 * 64];                    // 16KB: mask rows
  __shared__ __align__(16) _Float16 klds[2][32 * 32];   // 2 x 2KB: K tiles
  __shared__ __align__(16) _Float16 vlds[2][32 * 32];   // 2 x 2KB: V tiles
  __shared__ __align__(16) _Float16 plds[4][16 * 32];   // per-wave P tiles

  int tid = threadIdx.x;            // 0..127
  int lane = tid & 31;
  int w = tid >> 5;                 // wave 0..3
  int qb = blockIdx.x;              // 0..31  (2048 / 64)
  int bh = blockIdx.y;              // 0..63
  int b = bh >> 3, h = bh & 7;
  int hi = lane >> 4;
  const float NEG_INF = -__builtin_inff();

  // stage mask words for all 64 Q rows of this block
  const uint32_t* mrow = maskbits + ((size_t)b * N_ + qb * 64) * 64;
  for (int i = tid; i < 64 * 64; i += 128) mlds[i] = mrow[i];

  const _Float16* Qb = Qh + ((size_t)bh * N_ + qb * 64 + w * 16) * HD_;
  const _Float16* Kb = Kh + (size_t)bh * N_ * HD_;
  const _Float16* Vb = Vt + (size_t)bh * HD_ * N_;

  v16h qf = load_frag_A(Qb, HD_, lane);

  v8f o0 = {0.f, 0.f, 0.f, 0.f, 0.f, 0.f, 0.f, 0.f};
  v8f o1 = o0;
  float M[8], L[8];
#pragma unroll
  for (int r = 0; r < 8; ++r) { M[r] = NEG_INF; L[r] = 0.f; }

  const int NC = N_ / 32;  // 64 key chunks

  // prologue: stage chunk 0 into buffer 0 (K chunk is contiguous 2KB;
  // V chunk is 32 rows x 64B from the [hd][n] transposed layout)
  {
    uint4 kreg = ((const uint4*)Kb)[tid];
    uint4 vreg = *(const uint4*)(Vb + (size_t)(tid >> 2) * N_ + (tid & 3) * 8);
    ((uint4*)klds[0])[tid] = kreg;
    ((uint4*)vlds[0])[tid] = vreg;
  }

  for (int kc = 0; kc < NC; ++kc) {
    __syncthreads();  // staged K/V (and mask on kc==0) visible; plds WAR done
    int cur = kc & 1;

    // issue next chunk's global loads early (latency hidden by compute)
    uint4 kreg, vreg;
    if (kc + 1 < NC) {
      kreg = ((const uint4*)(Kb + (size_t)(kc + 1) * 32 * HD_))[tid];
      vreg = *(const uint4*)(Vb + (size_t)(tid >> 2) * N_ + (kc + 1) * 32 +
                             (tid & 3) * 8);
    }

    // ---- scores: S = Q K^T for two 16-key sub-tiles
    v16h kf0 = load_frag_B(klds[cur], HD_, lane);
    v16h kf1 = load_frag_B(klds[cur] + 16 * HD_, HD_, lane);
    v8f z8 = {0.f, 0.f, 0.f, 0.f, 0.f, 0.f, 0.f, 0.f};
    v8f c0 = wmma16(qf, kf0, z8);
    v8f c1 = wmma16(qf, kf1, z8);

    float s0[8], s1[8];
#pragma unroll
    for (int r = 0; r < 8; ++r) {
      uint32_t wd = mlds[(w * 16 + r + 8 * hi) * 64 + kc];
      s0[r] = ((wd >> (lane & 15)) & 1u) ? c0[r] * SCALE_ : NEG_INF;
      s1[r] = ((wd >> (16 + (lane & 15))) & 1u) ? c1[r] * SCALE_ : NEG_INF;
    }

    // ---- online softmax (row reductions across the 16-lane halves)
#pragma unroll
    for (int r = 0; r < 8; ++r) {
      float t = fmaxf(s0[r], s1[r]);
#pragma unroll
      for (int off = 1; off < 16; off <<= 1) t = fmaxf(t, __shfl_xor(t, off, 32));
      float Mn = fmaxf(M[r], t);
      float alpha = (Mn == NEG_INF) ? 1.f : __expf(M[r] - Mn);  // exp(-inf)=0
      float p0 = (s0[r] > -3.0e38f) ? __expf(s0[r] - Mn) : 0.f;
      float p1 = (s1[r] > -3.0e38f) ? __expf(s1[r] - Mn) : 0.f;
      float rs = p0 + p1;
#pragma unroll
      for (int off = 1; off < 16; off <<= 1) rs += __shfl_xor(rs, off, 32);
      L[r] = L[r] * alpha + rs;
      M[r] = Mn;
      o0[r] *= alpha;
      o1[r] *= alpha;
      int m = r + 8 * hi, col = lane & 15;
      plds[w][m * 32 + col] = (_Float16)p0;
      plds[w][m * 32 + 16 + col] = (_Float16)p1;
    }
    __syncthreads();  // P tiles visible; K-tile ds reads complete

    // ---- O += P V  (V fragments from the shared LDS tile)
    v16h pf = load_frag_A(plds[w], 32, lane);
    v16h vf0 = load_frag_B(vlds[cur], 32, lane);             // hd 0..15
    v16h vf1 = load_frag_B(vlds[cur] + 16 * 32, 32, lane);   // hd 16..31
    o0 = wmma16(pf, vf0, o0);
    o1 = wmma16(pf, vf1, o1);

    // ---- park next chunk into the other buffer
    if (kc + 1 < NC) {
      int nxt = cur ^ 1;
      ((uint4*)klds[nxt])[tid] = kreg;
      ((uint4*)vlds[nxt])[tid] = vreg;
    }
  }

  // ---- normalize and store (fully-masked row -> 0, matches nan_to_num)
  int col = lane & 15;
#pragma unroll
  for (int r = 0; r < 8; ++r) {
    float inv = (L[r] > 0.f) ? 1.f / L[r] : 0.f;
    int n = qb * 64 + w * 16 + r + 8 * hi;
    size_t rowoff = ((size_t)b * N_ + n) * DOUT_ + h * HD_;
    Oh[rowoff + col] = (_Float16)(o0[r] * inv);
    Oh[rowoff + 16 + col] = (_Float16)(o1[r] * inv);
  }
}

// ---------------------------------------------------------------------------
// Phase 3: output projection out = Oh Wo^T + bo, f32 result.
// ---------------------------------------------------------------------------
__global__ __launch_bounds__(32) void outproj_kernel(
    const _Float16* __restrict__ Oh, const _Float16* __restrict__ Woh,
    const float* __restrict__ bo, float* __restrict__ out) {
  int lane = threadIdx.x & 31;
  int rowblk = blockIdx.x;  // 0..1023
  int ot = blockIdx.y;      // 0..15

  const _Float16* Abase = Oh + (size_t)rowblk * 16 * DOUT_;
  const _Float16* Bbase = Woh + (size_t)ot * 16 * DOUT_;

  v8f c = {0.f, 0.f, 0.f, 0.f, 0.f, 0.f, 0.f, 0.f};
#pragma unroll
  for (int kc = 0; kc < DOUT_ / 32; ++kc) {
    v16h a = load_frag_A(Abase + kc * 32, DOUT_, lane);
    v16h b = load_frag_B(Bbase + kc * 32, DOUT_, lane);
    c = wmma16(a, b, c);
  }

  int hi = lane >> 4;
  int o = ot * 16 + (lane & 15);
  float bs = bo[o];
#pragma unroll
  for (int r = 0; r < 8; ++r) {
    size_t row = (size_t)rowblk * 16 + r + 8 * hi;
    out[row * DOUT_ + o] = c[r] + bs;
  }
}

// ---------------------------------------------------------------------------
extern "C" void kernel_launch(void* const* d_in, const int* in_sizes, int n_in,
                              void* d_out, int out_size, void* d_ws, size_t ws_size,
                              hipStream_t stream) {
  (void)in_sizes; (void)n_in; (void)out_size; (void)ws_size;
  const float* x  = (const float*)d_in[0];
  const int*  adj = (const int*)d_in[1];
  const float* Wq = (const float*)d_in[2];
  const float* bq = (const float*)d_in[3];
  const float* Wk = (const float*)d_in[4];
  const float* bk = (const float*)d_in[5];
  const float* Wv = (const float*)d_in[6];
  const float* bv = (const float*)d_in[7];
  const float* Wo = (const float*)d_in[8];
  const float* bo = (const float*)d_in[9];
  float* out = (float*)d_out;

  char* ws = (char*)d_ws;
  size_t off = 0;
  auto alloc = [&](size_t bytes) -> void* {
    void* p = ws + off;
    off += (bytes + 255) & ~(size_t)255;
    return p;
  };
  const size_t nBN = (size_t)B_ * N_;  // 16384
  _Float16* xh  = (_Float16*)alloc(nBN * DIN_ * 2);            // 8 MB
  _Float16* Wqh = (_Float16*)alloc((size_t)DOUT_ * DIN_ * 2);  // 128 KB each
  _Float16* Wkh = (_Float16*)alloc((size_t)DOUT_ * DIN_ * 2);
  _Float16* Wvh = (_Float16*)alloc((size_t)DOUT_ * DIN_ * 2);
  _Float16* Woh = (_Float16*)alloc((size_t)DOUT_ * DOUT_ * 2);
  _Float16* Qh  = (_Float16*)alloc(nBN * DOUT_ * 2);           // 8 MB
  _Float16* Kh  = (_Float16*)alloc(nBN * DOUT_ * 2);
  _Float16* Vt  = (_Float16*)alloc(nBN * DOUT_ * 2);
  _Float16* Oh  = (_Float16*)alloc(nBN * DOUT_ * 2);
  uint32_t* mbits = (uint32_t*)alloc(nBN * 64 * 4);            // 4 MB

  // Phase 0: conversions + adjacency bit-pack
  int nx = (int)(nBN * DIN_);
  cvt_f16_kernel<<<(nx + 255) / 256, 256, 0, stream>>>(x, xh, nx);
  int nw = DOUT_ * DIN_;
  cvt_f16_kernel<<<(nw + 255) / 256, 256, 0, stream>>>(Wq, Wqh, nw);
  cvt_f16_kernel<<<(nw + 255) / 256, 256, 0, stream>>>(Wk, Wkh, nw);
  cvt_f16_kernel<<<(nw + 255) / 256, 256, 0, stream>>>(Wv, Wvh, nw);
  cvt_f16_kernel<<<(nw + 255) / 256, 256, 0, stream>>>(Wo, Woh, nw);
  // total adj words = 8*2048*64 = 1,048,576; 32 words/wave, 8 waves/block
  pack_adj_kernel<<<4096, 256, 0, stream>>>(adj, mbits);

  // Phase 1: Q/K/V projections via WMMA
  proj_kernel<<<dim3(1024, 16, 3), 32, 0, stream>>>(xh, Wqh, Wkh, Wvh, bq, bk, bv,
                                                    Qh, Kh, Vt);

  // Phase 2: masked flash attention (4 waves / 64 Q rows per block)
  attn_kernel<<<dim3(N_ / 64, B_ * H_), 128, 0, stream>>>(Qh, Kh, Vt, mbits, Oh);

  // Phase 3: output projection
  outproj_kernel<<<dim3(1024, 16), 32, 0, stream>>>(Oh, Woh, bo, out);
}
